// RSSM_83391085019293
// MI455X (gfx1250) — matrix operand order repaired and checked by reading, not compiled, not run on hardware
//
#include <hip/hip_runtime.h>

// ---------------------------------------------------------------------------
// RSSM rollout for MI455X (gfx1250, wave32, WMMA).
// GEMMs: bf16 WMMA (v_wmma_f32_16x16x32_bf16), f32 accumulation.
// Weight tiles staged global->LDS with CDNA5 async copies (ASYNCcnt),
// double-buffered; steady-state loop is branch-free (last iter peeled).
// Wave tile 32x64: each LDS B-fragment feeds two WMMAs (1:1 wmma:ds ratio).
// ---------------------------------------------------------------------------

#define B_    256
#define T_    64
#define D_    1024
#define SD_   32
#define SC_   32
#define ADIM_ 6
#define KPRE  1056   // 1030 (SD*SC + ADIM) padded up to a multiple of 32

#define WROW  40     // LDS row stride in bf16 (80B): 32 data + 8 pad (bank spread)

typedef __bf16 bf16;
typedef __attribute__((ext_vector_type(16))) __bf16 v16bf;
typedef __attribute__((ext_vector_type(8)))  __bf16 v8bf;
typedef __attribute__((ext_vector_type(8)))  float  v8f;

static __device__ __forceinline__ bf16 f2bf(float f) {
  unsigned u = __float_as_uint(f);
  unsigned r = u + 0x7FFFu + ((u >> 16) & 1u);   // round-to-nearest-even
  unsigned short h = (unsigned short)(r >> 16);
  return __builtin_bit_cast(bf16, h);
}

// One 16B async DMA: global -> LDS (tracked by ASYNCcnt, not LOADcnt).
static __device__ __forceinline__ void async_copy16(unsigned lds_off,
                                                    const bf16* gsrc) {
  unsigned long long ga = (unsigned long long)gsrc;
  asm volatile("global_load_async_to_lds_b128 %0, %1, off"
               :: "v"(lds_off), "v"(ga) : "memory");
}

static __device__ __forceinline__ v16bf ld_frag_lds(const bf16* p) {
  v8bf lo = *(const v8bf*)p;
  v8bf hi = *(const v8bf*)(p + 8);
  return __builtin_shufflevector(lo, hi,
      0,1,2,3,4,5,6,7,8,9,10,11,12,13,14,15);
}

// ---------------------------------------------------------------------------
// GEMM: C[M=256, N] = A[256, K](bf16, row stride lda) * W[N, K]^T + bias
// Block: 256 threads = 8 waves; block tile 256(M) x 64(N); wave tile 32x64.
// Weight tile (64 x 32 bf16 = 4KB) staged in LDS, double-buffered:
//   steady state: issue async for buf^1, wait asynccnt<=1 (in-order => buf
//   landed), barrier, 8 ds-read frags + 8 WMMAs, barrier.  Last iter peeled.
// Fragment layouts used here (wave32; A and B both contiguous k):
//   A 16x32: lane<16 -> m=lane,   K = k0..k0+15
//            lane>=16 -> m=lane-16, K = k0+16..k0+31
//   B 32x16: lane -> n = lane&15, K = k0 + (lane<16 ? 0 : 16) .. +15
//   C 16x16: vgpr v -> m = v + (lane&16 ? 8 : 0), n = lane&15
// ---------------------------------------------------------------------------
__global__ __launch_bounds__(256)
void gemm_bf16_wmma(const bf16* __restrict__ A, int lda,
                    const bf16* __restrict__ W,
                    const float* __restrict__ bias,
                    float* __restrict__ C, int N, int K)
{
  __shared__ bf16 wtile[2][64 * WROW];

  const int tid   = threadIdx.x;
  const int lane  = tid & 31;
  const int wave  = tid >> 5;
  const int mbase = wave * 32;             // wave owns rows [mbase, mbase+32)
  const int nblk  = blockIdx.x * 64;

  // --- async-copy assignment: 4 threads per weight row, 16B each -----------
  const int lr = tid >> 2;                 // 0..63: row within N-tile
  const int lq = tid & 3;                  // 16B chunk within the 64B row
  const bf16* gsrc = W + (size_t)(nblk + lr) * K + lq * 8;
  const unsigned lds_my = (unsigned)(lr * WROW + lq * 8) * 2u; // byte offset
  const unsigned lds_b[2] = {
    (unsigned)(unsigned long long)(&wtile[0][0]),
    (unsigned)(unsigned long long)(&wtile[1][0])
  };

  // --- per-lane fragment coordinates ---------------------------------------
  const int aoff = (lane & 16) ? 16 : 0;
  const int boff = (lane & 16) ? 16 : 0;
  const int ncol = lane & 15;
  const bf16* arow0 = A + (size_t)(mbase + (lane & 15)) * lda;       // m-tile 0
  const bf16* arow1 = arow0 + (size_t)16 * lda;                      // m-tile 1

  v8f acc[2][4] = {};

  // prologue: stage first weight tile
  async_copy16(lds_b[0] + lds_my, gsrc);

  int buf = 0, k0 = 0;
  for (; k0 < K - 32; k0 += 32, buf ^= 1) {   // steady state: branch-free
    async_copy16(lds_b[buf ^ 1] + lds_my, gsrc + (k0 + 32));

    v16bf a0 = ld_frag_lds(arow0 + k0 + aoff);   // per-wave A (global, L2)
    v16bf a1 = ld_frag_lds(arow1 + k0 + aoff);

    asm volatile("s_wait_asynccnt 0x1" ::: "memory");
    __syncthreads();                        // tile visible to all 8 waves
    const bf16* wt = &wtile[buf][0];
#pragma unroll
    for (int j = 0; j < 4; ++j) {
      v16bf b = ld_frag_lds(wt + (j * 16 + ncol) * WROW + boff);
      acc[0][j] = __builtin_amdgcn_wmma_f32_16x16x32_bf16(
          false, a0, false, b, (short)0, acc[0][j], false, false);
      acc[1][j] = __builtin_amdgcn_wmma_f32_16x16x32_bf16(
          false, a1, false, b, (short)0, acc[1][j], false, false);
    }
    __syncthreads();                        // all reads done before overwrite
  }
  // peeled final iteration (no further async issue)
  {
    v16bf a0 = ld_frag_lds(arow0 + k0 + aoff);
    v16bf a1 = ld_frag_lds(arow1 + k0 + aoff);
    asm volatile("s_wait_asynccnt 0x0" ::: "memory");
    __syncthreads();
    const bf16* wt = &wtile[buf][0];
#pragma unroll
    for (int j = 0; j < 4; ++j) {
      v16bf b = ld_frag_lds(wt + (j * 16 + ncol) * WROW + boff);
      acc[0][j] = __builtin_amdgcn_wmma_f32_16x16x32_bf16(
          false, a0, false, b, (short)0, acc[0][j], false, false);
      acc[1][j] = __builtin_amdgcn_wmma_f32_16x16x32_bf16(
          false, a1, false, b, (short)0, acc[1][j], false, false);
    }
  }

  const int radd = (lane & 16) ? 8 : 0;
#pragma unroll
  for (int mi = 0; mi < 2; ++mi) {
#pragma unroll
    for (int j = 0; j < 4; ++j) {
      const int col = nblk + j * 16 + ncol;
      const float bb = bias ? bias[col] : 0.0f;
#pragma unroll
      for (int v = 0; v < 8; ++v) {
        const int row = mbase + mi * 16 + radd + v;
        C[(size_t)row * N + col] = acc[mi][j][v] + bb;
      }
    }
  }
}

// ---- LayerNorm(row of 1024) + SiLU -> bf16 pack ---------------------------
__global__ __launch_bounds__(256)
void ln_silu_pack(const float* __restrict__ X,
                  const float* __restrict__ g, const float* __restrict__ b,
                  bf16* __restrict__ out, int ldo)
{
  const int row = blockIdx.x;
  const float* x = X + (size_t)row * D_;
  float v[4], s = 0.f, s2 = 0.f;
#pragma unroll
  for (int i = 0; i < 4; ++i) {
    v[i] = x[threadIdx.x + 256 * i];
    s += v[i]; s2 += v[i] * v[i];
  }
  __shared__ float sa[8], sb[8];
  for (int off = 16; off; off >>= 1) {
    s  += __shfl_xor(s,  off, 32);
    s2 += __shfl_xor(s2, off, 32);
  }
  if ((threadIdx.x & 31) == 0) { sa[threadIdx.x >> 5] = s; sb[threadIdx.x >> 5] = s2; }
  __syncthreads();
  s = 0.f; s2 = 0.f;
#pragma unroll
  for (int i = 0; i < 8; ++i) { s += sa[i]; s2 += sb[i]; }
  const float mean = s * (1.0f / 1024.0f);
  const float var  = s2 * (1.0f / 1024.0f) - mean * mean;
  const float rstd = rsqrtf(var + 1e-5f);
#pragma unroll
  for (int i = 0; i < 4; ++i) {
    const int j = threadIdx.x + 256 * i;
    const float y = (v[i] - mean) * rstd * g[j] + b[j];
    out[(size_t)row * ldo + j] = f2bf(y / (1.0f + expf(-y)));
  }
}

// ---- GRU gate combine + LayerNorm + pack [h | e] for next GEMMs -----------
__global__ __launch_bounds__(256)
void gru_combine(const float* __restrict__ gx, const float* __restrict__ gh,
                 float* __restrict__ h,
                 const float* __restrict__ gg, const float* __restrict__ gb,
                 const float* __restrict__ embeds, int t,
                 float* __restrict__ hs_out, bf16* __restrict__ he_bf)
{
  const int bb = blockIdx.x;
  float hv[4], s = 0.f, s2 = 0.f;
#pragma unroll
  for (int i = 0; i < 4; ++i) {
    const int j = threadIdx.x + 256 * i;
    const size_t g3 = (size_t)bb * 3072;
    const float xr = gx[g3 + j], xu = gx[g3 + 1024 + j], xn = gx[g3 + 2048 + j];
    const float hr = gh[g3 + j], hu = gh[g3 + 1024 + j], hn = gh[g3 + 2048 + j];
    const float r = 1.f / (1.f + expf(-(xr + hr)));
    const float u = 1.f / (1.f + expf(-(xu + hu)));
    const float n = tanhf(xn + r * hn);
    const float hp = h[(size_t)bb * 1024 + j];
    hv[i] = (1.f - u) * n + u * hp;
    s += hv[i]; s2 += hv[i] * hv[i];
  }
  __shared__ float sa[8], sb[8];
  for (int off = 16; off; off >>= 1) {
    s  += __shfl_xor(s,  off, 32);
    s2 += __shfl_xor(s2, off, 32);
  }
  if ((threadIdx.x & 31) == 0) { sa[threadIdx.x >> 5] = s; sb[threadIdx.x >> 5] = s2; }
  __syncthreads();
  s = 0.f; s2 = 0.f;
#pragma unroll
  for (int i = 0; i < 8; ++i) { s += sa[i]; s2 += sb[i]; }
  const float mean = s * (1.0f / 1024.0f);
  const float var  = s2 * (1.0f / 1024.0f) - mean * mean;
  const float rstd = rsqrtf(var + 1e-5f);
#pragma unroll
  for (int i = 0; i < 4; ++i) {
    const int j = threadIdx.x + 256 * i;
    const float y = (hv[i] - mean) * rstd * gg[j] + gb[j];
    h[(size_t)bb * 1024 + j] = y;
    hs_out[((size_t)bb * T_ + t) * 1024 + j] = y;
    he_bf[(size_t)bb * 2048 + j] = f2bf(y);
    he_bf[(size_t)bb * 2048 + 1024 + j] =
        f2bf(embeds[((size_t)bb * T_ + t) * 1024 + j]);
  }
}

// ---- softmax(32) + unimix + (optional) Gumbel argmax -> one-hot z ---------
// One wave32 per (b, d) group; lane == class index.
__global__ __launch_bounds__(256)
void softmax_sample(const float* __restrict__ logits,
                    const float* __restrict__ u, int t,
                    float* __restrict__ p_out,
                    float* __restrict__ z_out, bf16* __restrict__ za)
{
  const int gid = blockIdx.x * 8 + (threadIdx.x >> 5);
  const int b = gid >> 5, d = gid & 31, c = threadIdx.x & 31;
  const float lg = logits[(size_t)b * 1024 + d * 32 + c];
  float mx = lg;
  for (int off = 16; off; off >>= 1) mx = fmaxf(mx, __shfl_xor(mx, off, 32));
  const float e = expf(lg - mx);
  float sum = e;
  for (int off = 16; off; off >>= 1) sum += __shfl_xor(sum, off, 32);
  const float p = 0.99f * (e / sum) + 0.01f / 32.0f;
  const size_t oidx = (((size_t)b * T_ + t) * 32 + d) * 32 + c;
  p_out[oidx] = p;
  if (z_out != nullptr) {
    const float uu = u[(((size_t)t * B_ + b) * 32 + d) * 32 + c];
    float sc = logf(p) + (-logf(-logf(uu)));
    int idx = c;
    for (int off = 16; off; off >>= 1) {
      const float os = __shfl_xor(sc, off, 32);
      const int   oi = __shfl_xor(idx, off, 32);
      if (os > sc || (os == sc && oi < idx)) { sc = os; idx = oi; }
    }
    const float oh = (c == idx) ? 1.0f : 0.0f;
    const float z = (oh + p) - p;          // straight-through, fwd == one-hot
    z_out[oidx] = z;
    za[(size_t)b * KPRE + d * 32 + c] = f2bf(z);
  }
}

// ---- small packers / init -------------------------------------------------
__global__ void pack_actions(const float* __restrict__ act, int t,
                             bf16* __restrict__ za)
{
  const int i = blockIdx.x * 256 + threadIdx.x;
  if (i < B_ * ADIM_) {
    const int bb = i / ADIM_, k = i % ADIM_;
    za[(size_t)bb * KPRE + 1024 + k] =
        f2bf(act[((size_t)bb * T_ + t) * ADIM_ + k]);
  }
}

__global__ void init_state(const float* __restrict__ h0,
                           const float* __restrict__ z0,
                           float* __restrict__ h, bf16* __restrict__ he_bf,
                           bf16* __restrict__ za)
{
  const int bb = blockIdx.x;
  for (int j = threadIdx.x; j < KPRE; j += 256) {
    const float zv = (j < 1024) ? z0[(size_t)bb * 1024 + j] : 0.0f;
    za[(size_t)bb * KPRE + j] = f2bf(zv);
  }
  for (int j = threadIdx.x; j < 1024; j += 256) {
    const float hv = h0[(size_t)bb * 1024 + j];
    h[(size_t)bb * 1024 + j] = hv;
    he_bf[(size_t)bb * 2048 + j] = f2bf(hv);
    he_bf[(size_t)bb * 2048 + 1024 + j] = f2bf(0.0f);
  }
}

__global__ void convert_weight(const float* __restrict__ in,
                               bf16* __restrict__ out,
                               int N, int Kin, int Kout)
{
  const size_t idx = (size_t)blockIdx.x * 256 + threadIdx.x;
  if (idx < (size_t)N * Kout) {
    const int n = (int)(idx / Kout), k = (int)(idx % Kout);
    out[idx] = (k < Kin) ? f2bf(in[(size_t)n * Kin + k]) : f2bf(0.0f);
  }
}

// ---------------------------------------------------------------------------
extern "C" void kernel_launch(void* const* d_in, const int* in_sizes, int n_in,
                              void* d_out, int out_size, void* d_ws,
                              size_t ws_size, hipStream_t stream)
{
  (void)in_sizes; (void)n_in; (void)out_size; (void)ws_size;
  const float* embeds  = (const float*)d_in[0];
  const float* actions = (const float*)d_in[1];
  const float* h_init  = (const float*)d_in[2];
  const float* z_init  = (const float*)d_in[3];
  const float* W_pre   = (const float*)d_in[4];
  const float* b_pre   = (const float*)d_in[5];
  const float* g_pre   = (const float*)d_in[6];
  const float* be_pre  = (const float*)d_in[7];
  const float* W_ih    = (const float*)d_in[8];
  const float* W_hh    = (const float*)d_in[9];
  const float* b_ih    = (const float*)d_in[10];
  const float* b_hh    = (const float*)d_in[11];
  const float* g_gn    = (const float*)d_in[12];
  const float* b_gn    = (const float*)d_in[13];
  const float* Wp1     = (const float*)d_in[14];
  const float* bp1     = (const float*)d_in[15];
  const float* gp      = (const float*)d_in[16];
  const float* bpn     = (const float*)d_in[17];
  const float* Wp2     = (const float*)d_in[18];
  const float* bp2     = (const float*)d_in[19];
  const float* Wq1     = (const float*)d_in[20];
  const float* bq1     = (const float*)d_in[21];
  const float* gq      = (const float*)d_in[22];
  const float* bqn     = (const float*)d_in[23];
  const float* Wq2     = (const float*)d_in[24];
  const float* bq2     = (const float*)d_in[25];
  const float* u_prior = (const float*)d_in[26];
  const float* u_post  = (const float*)d_in[27];

  float* out = (float*)d_out;
  const size_t OSZ = (size_t)B_ * T_ * 1024;
  float* hs = out;
  float* zs = out + OSZ;
  float* pp = out + 2 * OSZ;
  float* qp = out + 3 * OSZ;

  char* ws = (char*)d_ws;
  size_t off = 0;
  auto take = [&](size_t bytes) -> void* {
    void* p = ws + off;
    off += (bytes + 255) & ~(size_t)255;
    return p;
  };
  bf16* Wpre_bf = (bf16*)take((size_t)1024 * KPRE * 2);
  bf16* Wih_bf  = (bf16*)take((size_t)3072 * 1024 * 2);
  bf16* Whh_bf  = (bf16*)take((size_t)3072 * 1024 * 2);
  bf16* Wp1_bf  = (bf16*)take((size_t)1024 * 1024 * 2);
  bf16* Wp2_bf  = (bf16*)take((size_t)1024 * 1024 * 2);
  bf16* Wq1_bf  = (bf16*)take((size_t)1024 * 2048 * 2);
  bf16* Wq2_bf  = (bf16*)take((size_t)1024 * 1024 * 2);
  bf16* za      = (bf16*)take((size_t)B_ * KPRE * 2);
  bf16* x_bf    = (bf16*)take((size_t)B_ * 1024 * 2);
  bf16* he_bf   = (bf16*)take((size_t)B_ * 2048 * 2);   // [h | e]
  bf16* t1_bf   = (bf16*)take((size_t)B_ * 1024 * 2);
  float* h_f32  = (float*)take((size_t)B_ * 1024 * 4);
  float* pre_raw= (float*)take((size_t)B_ * 1024 * 4);
  float* gx_raw = (float*)take((size_t)B_ * 3072 * 4);
  float* gh_raw = (float*)take((size_t)B_ * 3072 * 4);
  float* t1_raw = (float*)take((size_t)B_ * 1024 * 4);
  float* t2_raw = (float*)take((size_t)B_ * 1024 * 4);

  auto cvt = [&](const float* src, bf16* dst, int N, int Kin, int Kout) {
    const size_t tot = (size_t)N * Kout;
    convert_weight<<<(unsigned)((tot + 255) / 256), 256, 0, stream>>>(
        src, dst, N, Kin, Kout);
  };
  cvt(W_pre, Wpre_bf, 1024, 1030, KPRE);
  cvt(W_ih,  Wih_bf,  3072, 1024, 1024);
  cvt(W_hh,  Whh_bf,  3072, 1024, 1024);
  cvt(Wp1,   Wp1_bf,  1024, 1024, 1024);
  cvt(Wp2,   Wp2_bf,  1024, 1024, 1024);
  cvt(Wq1,   Wq1_bf,  1024, 2048, 2048);
  cvt(Wq2,   Wq2_bf,  1024, 1024, 1024);

  init_state<<<B_, 256, 0, stream>>>(h_init, z_init, h_f32, he_bf, za);

  for (int t = 0; t < T_; ++t) {
    pack_actions<<<6, 256, 0, stream>>>(actions, t, za);
    // x = silu(LN([z,a] @ Wpre^T + b))
    gemm_bf16_wmma<<<16, 256, 0, stream>>>(za, KPRE, Wpre_bf, b_pre,
                                           pre_raw, 1024, KPRE);
    ln_silu_pack<<<B_, 256, 0, stream>>>(pre_raw, g_pre, be_pre, x_bf, 1024);
    // gx = x @ Wih^T + b_ih ; gh = h_{t-1} @ Whh^T + b_hh
    gemm_bf16_wmma<<<48, 256, 0, stream>>>(x_bf, 1024, Wih_bf, b_ih,
                                           gx_raw, 3072, 1024);
    gemm_bf16_wmma<<<48, 256, 0, stream>>>(he_bf, 2048, Whh_bf, b_hh,
                                           gh_raw, 3072, 1024);
    gru_combine<<<B_, 256, 0, stream>>>(gx_raw, gh_raw, h_f32, g_gn, b_gn,
                                        embeds, t, hs, he_bf);
    // prior MLP + sample (probs only)
    gemm_bf16_wmma<<<16, 256, 0, stream>>>(he_bf, 2048, Wp1_bf, bp1,
                                           t1_raw, 1024, 1024);
    ln_silu_pack<<<B_, 256, 0, stream>>>(t1_raw, gp, bpn, t1_bf, 1024);
    gemm_bf16_wmma<<<16, 256, 0, stream>>>(t1_bf, 1024, Wp2_bf, bp2,
                                           t2_raw, 1024, 1024);
    softmax_sample<<<1024, 256, 0, stream>>>(t2_raw, u_prior, t, pp,
                                             nullptr, nullptr);
    // posterior MLP + sample (probs + one-hot z feeding next step)
    gemm_bf16_wmma<<<16, 256, 0, stream>>>(he_bf, 2048, Wq1_bf, bq1,
                                           t1_raw, 1024, 2048);
    ln_silu_pack<<<B_, 256, 0, stream>>>(t1_raw, gq, bqn, t1_bf, 1024);
    gemm_bf16_wmma<<<16, 256, 0, stream>>>(t1_bf, 1024, Wq2_bf, bq2,
                                           t2_raw, 1024, 1024);
    softmax_sample<<<1024, 256, 0, stream>>>(t2_raw, u_post, t, qp, zs, za);
  }
}
